// MirrorLM_73143293051251
// MI455X (gfx1250) — matrix operand order repaired and checked by reading, not compiled, not run on hardware
//
#include <hip/hip_runtime.h>

// ---------------- types ----------------
typedef __bf16 bf16_t;
typedef __attribute__((ext_vector_type(16))) __bf16        v16bf;
typedef __attribute__((ext_vector_type(8)))  float         v8f;
typedef __attribute__((ext_vector_type(4)))  unsigned int  v4u;
typedef __attribute__((ext_vector_type(8)))  int           v8i;
typedef __attribute__((ext_vector_type(4)))  int           v4i;
typedef __attribute__((ext_vector_type(4)))  unsigned int  u32x4;

union FragAB { v16bf v; u32x4 u[2]; };

static __device__ __forceinline__ v8f wmma_bf16(v16bf a, v16bf b, v8f c) {
  return __builtin_amdgcn_wmma_f32_16x16x32_bf16(false, a, false, b, (short)0, c, false, false);
}

#define BB   32
#define TT   2048
#define DD   512
#define HH   2
#define HDm  256
#define HIDm 512
#define VV   50257

// ---------------- helpers ----------------
__global__ void k_transpose_bf16(const float* __restrict__ W, bf16_t* __restrict__ Wt,
                                 int K, int N) {
  size_t i = (size_t)blockIdx.x * 256 + threadIdx.x;
  if (i >= (size_t)K * N) return;
  int k = (int)(i / N), n = (int)(i % N);
  Wt[(size_t)n * K + k] = (bf16_t)W[i];
}

__global__ void k_embed(const int* __restrict__ ids, const float* __restrict__ emb,
                        bf16_t* __restrict__ x) {
  size_t i = (size_t)blockIdx.x * 256 + threadIdx.x;
  if (i >= (size_t)BB * TT * DD) return;
  size_t row = i / DD; int d = (int)(i % DD);
  x[i] = (bf16_t)emb[(size_t)ids[row] * DD + d];
}

// ---------------- generic bf16 WMMA GEMM: C(MxN) = A(MxK) * Bt(NxK)^T (+bias) ----------------
// block = 256 thr = 8 waves -> 64x64 tile; wave (wm 0..3, wn 0..1): 16 rows x 32 cols (2 n-tiles)
__global__ void k_gemm_bf16(const bf16_t* __restrict__ A, int lda, long long sAb, long long sAh,
                            const bf16_t* __restrict__ Bt, int ldb, long long sBb, long long sBh,
                            const float* __restrict__ bias,
                            bf16_t* __restrict__ C, int ldc, long long sCb, long long sCh,
                            int M, int N, int K, int Hdim) {
  int bz = blockIdx.z;
  long long bb = bz / Hdim, hh = bz % Hdim;
  A  += bb * sAb + hh * sAh;
  Bt += bb * sBb + hh * sBh;
  C  += bb * sCb + hh * sCh;

  int lane = threadIdx.x & 31;
  int wave = threadIdx.x >> 5;
  int wm = wave >> 1, wn = wave & 1;
  int m0 = blockIdx.y * 64 + wm * 16;
  int n0 = blockIdx.x * 64 + wn * 32;

  int r     = lane & 15;
  int koffA = (lane >> 4) * 8;    // A: K in {koffA..koffA+7} U {koffA+16..koffA+23}
  int koffB = (lane >> 4) * 8;    // B: K in {koffB..koffB+7} U {koffB+16..koffB+23}

  v8f acc0 = {}; v8f acc1 = {};
  for (int k0 = 0; k0 < K; k0 += 32) {
    FragAB a, b0, b1;
    const bf16_t* pa = A + (size_t)(m0 + r) * lda + k0 + koffA;
    a.u[0] = *(const u32x4*)pa;
    a.u[1] = *(const u32x4*)(pa + 16);
    const bf16_t* pb0 = Bt + (size_t)(n0 + r) * ldb + k0 + koffB;
    b0.u[0] = *(const u32x4*)pb0;
    b0.u[1] = *(const u32x4*)(pb0 + 16);
    const bf16_t* pb1 = pb0 + (size_t)16 * ldb;
    b1.u[0] = *(const u32x4*)pb1;
    b1.u[1] = *(const u32x4*)(pb1 + 16);
    __builtin_prefetch(pb0 + 32, 0, 0);   // global_prefetch_b8 of next K panel
    acc0 = wmma_bf16(a.v, b0.v, acc0);
    acc1 = wmma_bf16(a.v, b1.v, acc1);
  }

  int cn = lane & 15;
  int rb = (lane >> 4) * 8;
#pragma unroll
  for (int i = 0; i < 8; ++i) {
    size_t row = (size_t)(m0 + rb + i);
    int c0 = n0 + cn, c1 = n0 + 16 + cn;
    float v0 = acc0[i], v1 = acc1[i];
    if (bias) { v0 += bias[c0]; v1 += bias[c1]; }
    C[row * ldc + c0] = (bf16_t)v0;
    C[row * ldc + c1] = (bf16_t)v1;
  }
}

// ---------------- attention elementwise ----------------
__device__ __forceinline__ float phi_f(float x) { return x > 0.f ? x + 1.f : __expf(x); }

__global__ void k_phirope_q(bf16_t* __restrict__ q) {   // in-place on (b,t,h,d) layout
  size_t i = (size_t)blockIdx.x * 256 + threadIdx.x;
  if (i >= (size_t)BB * TT * HH * 128) return;
  int d = (int)(i & 127);
  int h = (int)((i >> 7) & (HH - 1));
  size_t bt = i >> 8;
  int t = (int)(bt & (TT - 1));
  size_t base = bt * DD + (size_t)h * HDm;
  float x1 = phi_f((float)q[base + d]);
  float x2 = phi_f((float)q[base + 128 + d]);
  float fr = __expf(-9.210340371976184f * (float)d * (1.f / 128.f));
  float ang = (float)t * fr;
  float sn = __sinf(ang), cs = __cosf(ang);
  q[base + d]       = (bf16_t)(x1 * cs - x2 * sn);
  q[base + 128 + d] = (bf16_t)(x1 * sn + x2 * cs);
}

__global__ void k_phirope_kT(const bf16_t* __restrict__ kin, bf16_t* __restrict__ kT) {
  size_t i = (size_t)blockIdx.x * 256 + threadIdx.x;
  if (i >= (size_t)BB * TT * HH * 128) return;
  int d = (int)(i & 127);
  int h = (int)((i >> 7) & (HH - 1));
  size_t bt = i >> 8;
  int t = (int)(bt & (TT - 1));
  size_t b = bt >> 11;
  size_t base = bt * DD + (size_t)h * HDm;
  float x1 = phi_f((float)kin[base + d]);
  float x2 = phi_f((float)kin[base + 128 + d]);
  float fr = __expf(-9.210340371976184f * (float)d * (1.f / 128.f));
  float ang = (float)t * fr;
  float sn = __sinf(ang), cs = __cosf(ang);
  size_t bh = b * HH + h;
  kT[(bh * HDm + d) * TT + t]         = (bf16_t)(x1 * cs - x2 * sn);
  kT[(bh * HDm + 128 + d) * TT + t]   = (bf16_t)(x1 * sn + x2 * cs);
}

__global__ void k_relayout_vT(const bf16_t* __restrict__ vin, bf16_t* __restrict__ vT) {
  size_t i = (size_t)blockIdx.x * 256 + threadIdx.x;
  if (i >= (size_t)BB * TT * DD) return;
  int e = (int)(i & 255);
  int h = (int)((i >> 8) & 1);
  size_t bt = i >> 9;
  int t = (int)(bt & (TT - 1));
  size_t b = bt >> 11;
  vT[((b * HH + h) * HDm + e) * TT + t] = vin[bt * DD + (size_t)h * HDm + e];
}

__global__ void k_sigmoid(bf16_t* __restrict__ g, size_t n) {
  size_t i = (size_t)blockIdx.x * 256 + threadIdx.x;
  if (i >= n) return;
  float f = (float)g[i];
  g[i] = (bf16_t)(1.f / (1.f + __expf(-f)));
}

__global__ void k_ksum(const bf16_t* __restrict__ kT, float* __restrict__ ksum) {
  int idx = blockIdx.x * 256 + threadIdx.x;
  if (idx >= BB * HH * HDm) return;
  const bf16_t* p = kT + (size_t)idx * TT;
  float s = 0.f;
  for (int t = 0; t < TT; ++t) s += (float)p[t];
  ksum[idx] = s;
}

__global__ void k_z(const bf16_t* __restrict__ q, const float* __restrict__ ksum,
                    float* __restrict__ z) {
  size_t idx = (size_t)blockIdx.x * 256 + threadIdx.x;
  if (idx >= (size_t)BB * TT * HH) return;
  int h = (int)(idx & 1);
  size_t bt = idx >> 1;
  size_t b = bt >> 11;
  const bf16_t* qp = q + bt * DD + (size_t)h * HDm;
  const float* kp = ksum + (b * HH + h) * HDm;
  float s = 0.f;
  for (int d = 0; d < HDm; ++d) s += (float)qp[d] * kp[d];
  z[idx] = fmaxf(s, 1e-6f);
}

__global__ void k_combine(const bf16_t* __restrict__ o, const float* __restrict__ z,
                          const bf16_t* __restrict__ gate, const int* __restrict__ ids,
                          const float* __restrict__ emb, bf16_t* __restrict__ x2) {
  size_t i = (size_t)blockIdx.x * 256 + threadIdx.x;
  if (i >= (size_t)BB * TT * DD) return;
  int dg = (int)(i & 511);
  size_t bt = i >> 9;
  int h = dg >> 8;
  float outv = (float)o[i] / z[bt * HH + h];
  float g = (float)gate[i];
  float xv = emb[(size_t)ids[bt] * DD + dg];
  x2[i] = (bf16_t)(g * outv + (1.f - g) * xv);
}

// ---------------- GRU scan: 1 workgroup, 32 waves, h in LDS, WMMA per step ----------------
__global__ void k_gru_scan(const bf16_t* __restrict__ xg, const bf16_t* __restrict__ whhT,
                           const float* __restrict__ b_hh,
                           float* __restrict__ out_m2, float* __restrict__ out_m1) {
  extern __shared__ char smem[];
  bf16_t* hs = (bf16_t*)smem;                 // 32 x 512
  bf16_t* hg = hs + (size_t)BB * HIDm;        // 32 x 1536
  int tid = threadIdx.x;                       // 1024 threads
  int lane = tid & 31, wave = tid >> 5;        // 32 waves
  for (int i = tid; i < BB * HIDm; i += 1024) hs[i] = (bf16_t)0.f;
  __syncthreads();

  int r     = lane & 15;
  int koffA = (lane >> 4) * 8;
  int koffB = (lane >> 4) * 8;
  int rb    = (lane >> 4) * 8;

#pragma unroll 1
  for (int t = 0; t < TT; ++t) {
    // hg = h @ w_hh   (M=32, N=1536, K=512) : 192 tile jobs, 6 per wave
#pragma unroll 1
    for (int job = wave * 6; job < wave * 6 + 6; ++job) {
      int mt = job / 96, nt = job % 96;
      int m0 = mt * 16, n0 = nt * 16;
      v8f acc = {};
      for (int k0 = 0; k0 < HIDm; k0 += 32) {
        FragAB a, b;
        const bf16_t* pa = hs + (size_t)(m0 + r) * HIDm + k0 + koffA;
        a.u[0] = *(const u32x4*)pa;
        a.u[1] = *(const u32x4*)(pa + 16);
        const bf16_t* pb = whhT + (size_t)(n0 + r) * HIDm + k0 + koffB;
        b.u[0] = *(const u32x4*)pb;
        b.u[1] = *(const u32x4*)(pb + 16);
        acc = wmma_bf16(a.v, b.v, acc);
      }
#pragma unroll
      for (int i = 0; i < 8; ++i)
        hg[(size_t)(m0 + rb + i) * 1536 + n0 + r] = (bf16_t)acc[i];
    }
    __syncthreads();

    // gates + state update
    for (int e = tid; e < BB * HIDm; e += 1024) {
      int b = e >> 9, j = e & 511;
      size_t xrow = ((size_t)b * TT + t) * 1536;
      float xr = (float)xg[xrow + j];
      float xz = (float)xg[xrow + 512 + j];
      float xn = (float)xg[xrow + 1024 + j];
      size_t hrow = (size_t)b * 1536;
      float hr = (float)hg[hrow + j]        + b_hh[j];
      float hz = (float)hg[hrow + 512 + j]  + b_hh[512 + j];
      float hn = (float)hg[hrow + 1024 + j] + b_hh[1024 + j];
      float rr = 1.f / (1.f + __expf(-(xr + hr)));
      float zz = 1.f / (1.f + __expf(-(xz + hz)));
      float nn = tanhf(xn + rr * hn);
      float hold = (float)hs[e];
      float hnew = (1.f - zz) * nn + zz * hold;
      hs[e] = (bf16_t)hnew;
      if (t == TT - 2) out_m2[e] = hnew;
      if (t == TT - 1) out_m1[e] = hnew;
    }
    __syncthreads();
  }
}

// ---------------- tail reductions ----------------
__global__ void k_prederr(const float* __restrict__ m1, const float* __restrict__ m2,
                          float* __restrict__ out) {
  __shared__ float red[256];
  float s = 0.f;
  for (int i = threadIdx.x; i < BB * HIDm; i += 256) {
    float d = m1[i] - m2[i]; s += d * d;
  }
  red[threadIdx.x] = s; __syncthreads();
  for (int o = 128; o > 0; o >>= 1) {
    if (threadIdx.x < o) red[threadIdx.x] += red[threadIdx.x + o];
    __syncthreads();
  }
  if (threadIdx.x == 0) out[0] = sqrtf(red[0] / (float)(BB * HIDm));
}

__global__ void k_rmsnorm(const float* __restrict__ hn, const float* __restrict__ lnw,
                          float* __restrict__ h_out, bf16_t* __restrict__ h_bf) {
  __shared__ float red[256];
  __shared__ float inv;
  int b = blockIdx.x;
  const float* hp = hn + (size_t)b * HIDm;
  float s = 0.f;
  for (int j = threadIdx.x; j < HIDm; j += 256) { float v = hp[j]; s += v * v; }
  red[threadIdx.x] = s; __syncthreads();
  for (int o = 128; o > 0; o >>= 1) {
    if (threadIdx.x < o) red[threadIdx.x] += red[threadIdx.x + o];
    __syncthreads();
  }
  if (threadIdx.x == 0) inv = rsqrtf(red[0] / (float)HIDm + 1e-8f);
  __syncthreads();
  for (int j = threadIdx.x; j < HIDm; j += 256) {
    float v = lnw[j] * hp[j] * inv;
    h_out[(size_t)b * HIDm + j] = v;
    h_bf[(size_t)b * HIDm + j] = (bf16_t)v;
  }
}

// ---------------- head: logits = h(32x512) @ head_wT(Vx512)^T + b ; TDM stages weight tile ----
__global__ void k_head(const bf16_t* __restrict__ hb, const bf16_t* __restrict__ wT,
                       const float* __restrict__ head_b, float* __restrict__ logits) {
  extern __shared__ bf16_t ldsB[];   // 64 rows x 512 cols bf16 = 64KB
  int n0 = blockIdx.x * 64;

#if __has_builtin(__builtin_amdgcn_tensor_load_to_lds)
  if (threadIdx.x < 32) {
    unsigned long long ga = (unsigned long long)(const void*)(wT + (size_t)n0 * HIDm);
    unsigned lds_lo = (unsigned)(unsigned long long)(void*)&ldsB[0];
    int remain = VV - n0; if (remain > 64) remain = 64;
    v4u g0 = {0u, 0u, 0u, 0u};
    g0[0] = 1u;                                        // count = 1 (valid D#)
    g0[1] = lds_lo;                                    // lds_addr
    g0[2] = (unsigned)ga;                              // global_addr[31:0]
    g0[3] = (unsigned)((ga >> 32) & 0x01FFFFFFu) | (2u << 30);  // global_addr[56:32] | type=2
    v8i g1 = {0, 0, 0, 0, 0, 0, 0, 0};
    g1[0] = (1 << 16);                                 // data_size = 2 bytes
    g1[1] = (HIDm & 0xFFFF) << 16;                     // tensor_dim0 lo16 = 512
    g1[2] = ((remain & 0xFFFF) << 16);                 // tensor_dim0 hi=0 | tensor_dim1 lo16
    g1[3] = ((remain >> 16) & 0xFFFF) | (HIDm << 16);  // tensor_dim1 hi | tile_dim0 = 512
    g1[4] = 64;                                        // tile_dim1 = 64
    g1[5] = HIDm;                                      // tensor_dim0_stride lo32 = 512
    v4i gz = {0, 0, 0, 0};
    v8i gz8 = {0, 0, 0, 0, 0, 0, 0, 0};
    __builtin_amdgcn_tensor_load_to_lds(g0, g1, gz, gz, gz8, 0);
    __builtin_amdgcn_s_wait_tensorcnt(0);
  }
#else
  for (int i = threadIdx.x; i < 64 * HIDm; i += 256) {
    int rr_ = i >> 9, cc_ = i & 511;
    int gr = n0 + rr_;
    ldsB[i] = (gr < VV) ? wT[(size_t)gr * HIDm + cc_] : (bf16_t)0.f;
  }
#endif
  __syncthreads();

  int lane = threadIdx.x & 31, wave = threadIdx.x >> 5;  // 8 waves
  int mt = wave >> 2, nt = wave & 3;                     // 2 m-tiles x 4 n-tiles
  int m0 = mt * 16;
  int r = lane & 15;
  int koffA = (lane >> 4) * 8;
  int koffB = (lane >> 4) * 8;
  int rb = (lane >> 4) * 8;

  v8f acc = {};
  for (int k0 = 0; k0 < HIDm; k0 += 32) {
    FragAB a, b;
    const bf16_t* pa = hb + (size_t)(m0 + r) * HIDm + k0 + koffA;
    a.u[0] = *(const u32x4*)pa;
    a.u[1] = *(const u32x4*)(pa + 16);
    const bf16_t* pb = ldsB + (size_t)(nt * 16 + r) * HIDm + k0 + koffB;
    b.u[0] = *(const u32x4*)pb;
    b.u[1] = *(const u32x4*)(pb + 16);
    acc = wmma_bf16(a.v, b.v, acc);
  }
#pragma unroll
  for (int i = 0; i < 8; ++i) {
    int row = m0 + rb + i;
    int col = n0 + nt * 16 + r;
    if (col < VV) logits[(size_t)row * VV + col] = acc[i] + head_b[col];
  }
}

// ---------------- host ----------------
static void gemm(hipStream_t s,
                 const bf16_t* A, int lda, long long sAb, long long sAh,
                 const bf16_t* Bt, int ldb, long long sBb, long long sBh,
                 const float* bias, bf16_t* C, int ldc, long long sCb, long long sCh,
                 int M, int N, int K, int Z, int Hdim) {
  dim3 g(N / 64, M / 64, Z);
  k_gemm_bf16<<<g, 256, 0, s>>>(A, lda, sAb, sAh, Bt, ldb, sBb, sBh, bias,
                                C, ldc, sCb, sCh, M, N, K, Hdim);
}

static inline unsigned gsz(size_t n) { return (unsigned)((n + 255) / 256); }

extern "C" void kernel_launch(void* const* d_in, const int* in_sizes, int n_in,
                              void* d_out, int out_size, void* d_ws, size_t ws_size,
                              hipStream_t stream) {
  const int*   ids    = (const int*)  d_in[0];
  const float* emb    = (const float*)d_in[1];
  const float* wq     = (const float*)d_in[2];
  const float* wk     = (const float*)d_in[3];
  const float* wv     = (const float*)d_in[4];
  /* wo = d_in[5] is unused by the reference */
  const float* wg     = (const float*)d_in[6];
  const float* bg     = (const float*)d_in[7];
  const float* w_ih   = (const float*)d_in[8];
  const float* w_hh   = (const float*)d_in[9];
  const float* b_ih   = (const float*)d_in[10];
  const float* b_hh   = (const float*)d_in[11];
  const float* ln_w   = (const float*)d_in[12];
  const float* head_w = (const float*)d_in[13];
  const float* head_b = (const float*)d_in[14];

  float* logits = (float*)d_out;
  float* h_out  = logits + (size_t)BB * VV;
  float* pe_out = h_out + (size_t)BB * HIDm;

  char* wsb = (char*)d_ws;
  size_t off = 0;
  auto alloc = [&](size_t bytes) -> void* {
    void* p = wsb + off;
    off += (bytes + 255) & ~(size_t)255;
    return p;
  };
  const size_t NTD = (size_t)BB * TT * DD;         // 33.5M elems
  bf16_t* wqT  = (bf16_t*)alloc((size_t)DD * DD * 2);
  bf16_t* wkT  = (bf16_t*)alloc((size_t)DD * DD * 2);
  bf16_t* wvT  = (bf16_t*)alloc((size_t)DD * DD * 2);
  bf16_t* wgT  = (bf16_t*)alloc((size_t)DD * DD * 2);
  bf16_t* wihT = (bf16_t*)alloc((size_t)3 * HIDm * DD * 2);
  bf16_t* whhT = (bf16_t*)alloc((size_t)3 * HIDm * HIDm * 2);
  bf16_t* hwT  = (bf16_t*)alloc((size_t)VV * HIDm * 2);
  bf16_t* xb   = (bf16_t*)alloc(NTD * 2);
  bf16_t* qb   = (bf16_t*)alloc(NTD * 2);
  bf16_t* tmp  = (bf16_t*)alloc(NTD * 2);
  bf16_t* kT   = (bf16_t*)alloc(NTD * 2);
  bf16_t* vT   = (bf16_t*)alloc(NTD * 2);
  bf16_t* gate = (bf16_t*)alloc(NTD * 2);
  bf16_t* kvT  = (bf16_t*)alloc((size_t)BB * HH * HDm * HDm * 2);
  float*  ksum = (float*) alloc((size_t)BB * HH * HDm * 4);
  float*  zb   = (float*) alloc((size_t)BB * TT * HH * 4);
  bf16_t* xg   = (bf16_t*)alloc((size_t)BB * TT * 3 * HIDm * 2);
  float*  m2   = (float*) alloc((size_t)BB * HIDm * 4);
  float*  m1   = (float*) alloc((size_t)BB * HIDm * 4);
  bf16_t* hbf  = (bf16_t*)alloc((size_t)BB * HIDm * 2);

  // 1) weight transpose + bf16 convert
  k_transpose_bf16<<<gsz((size_t)DD * DD), 256, 0, stream>>>(wq, wqT, DD, DD);
  k_transpose_bf16<<<gsz((size_t)DD * DD), 256, 0, stream>>>(wk, wkT, DD, DD);
  k_transpose_bf16<<<gsz((size_t)DD * DD), 256, 0, stream>>>(wv, wvT, DD, DD);
  k_transpose_bf16<<<gsz((size_t)DD * DD), 256, 0, stream>>>(wg, wgT, DD, DD);
  k_transpose_bf16<<<gsz((size_t)DD * 3 * HIDm), 256, 0, stream>>>(w_ih, wihT, DD, 3 * HIDm);
  k_transpose_bf16<<<gsz((size_t)HIDm * 3 * HIDm), 256, 0, stream>>>(w_hh, whhT, HIDm, 3 * HIDm);
  k_transpose_bf16<<<gsz((size_t)DD * VV), 256, 0, stream>>>(head_w, hwT, DD, VV);

  // 2) embedding
  k_embed<<<gsz(NTD), 256, 0, stream>>>(ids, emb, xb);

  const int MT = BB * TT;  // 65536
  // 3) projections (WMMA GEMMs)
  gemm(stream, xb, DD,0,0, wqT, DD,0,0, nullptr, qb,  DD,0,0, MT, DD, DD, 1, 1);
  gemm(stream, xb, DD,0,0, wkT, DD,0,0, nullptr, tmp, DD,0,0, MT, DD, DD, 1, 1);
  k_phirope_q <<<gsz((size_t)BB * TT * HH * 128), 256, 0, stream>>>(qb);
  k_phirope_kT<<<gsz((size_t)BB * TT * HH * 128), 256, 0, stream>>>(tmp, kT);
  gemm(stream, xb, DD,0,0, wvT, DD,0,0, nullptr, tmp, DD,0,0, MT, DD, DD, 1, 1);
  k_relayout_vT<<<gsz(NTD), 256, 0, stream>>>(tmp, vT);
  gemm(stream, xb, DD,0,0, wgT, DD,0,0, bg, gate, DD,0,0, MT, DD, DD, 1, 1);
  k_sigmoid<<<gsz(NTD), 256, 0, stream>>>(gate, NTD);

  // 4) kvT[e,d] = sum_t v[t,e] k[t,d]   (A=vT, Bt=kT), per (b,h)
  gemm(stream, vT, TT, (long long)HDm * TT, 0,
               kT, TT, (long long)HDm * TT, 0,
       nullptr, kvT, HDm, (long long)HDm * HDm, 0,
       HDm, HDm, TT, BB * HH, 1);

  // 5) out[t,e] = sum_d q[t,d] kvT[e,d]  -> tmp, (b,t,h,e) layout
  gemm(stream, qb, DD, (long long)TT * DD, HDm,
               kvT, HDm, (long long)HH * HDm * HDm, (long long)HDm * HDm,
       nullptr, tmp, DD, (long long)TT * DD, HDm,
       TT, HDm, HDm, BB * HH, HH);

  // 6) z normalizer and gated combine -> x2 (reuse qb)
  k_ksum<<<gsz((size_t)BB * HH * HDm), 256, 0, stream>>>(kT, ksum);
  k_z<<<gsz((size_t)BB * TT * HH), 256, 0, stream>>>(qb, ksum, zb);
  k_combine<<<gsz(NTD), 256, 0, stream>>>(tmp, zb, gate, ids, emb, qb);

  // 7) xg = x2 @ w_ih + b_ih
  gemm(stream, qb, DD,0,0, wihT, DD,0,0, b_ih, xg, 3 * HIDm,0,0, MT, 3 * HIDm, DD, 1, 1);

  // 8) sequential GRU scan (persistent single workgroup, h in LDS, WMMA per step)
  size_t scan_lds = (size_t)BB * HIDm * 2 + (size_t)BB * 3 * HIDm * 2;  // 128KB
  k_gru_scan<<<1, 1024, scan_lds, stream>>>(xg, whhT, b_hh, m2, m1);

  // 9) tails
  k_prederr<<<1, 256, 0, stream>>>(m1, m2, pe_out);
  k_rmsnorm<<<BB, 256, 0, stream>>>(m1, ln_w, h_out, hbf);

  // 10) head GEMM with TDM-staged weight tiles
  k_head<<<(VV + 63) / 64, 256, (size_t)64 * HIDm * 2, stream>>>(hbf, hwT, head_b, logits);

  (void)in_sizes; (void)n_in; (void)out_size; (void)ws_size;
}
